// LSTM_Reverser_944892805644
// MI455X (gfx1250) — matrix operand order repaired and checked by reading, not compiled, not run on hardware
//
#include <hip/hip_runtime.h>
#include <hip/hip_bf16.h>

// ---------------------------------------------------------------------------
// Problem constants (from reference): T_IN=257, T_OUT=256, B=256, D=131,
// H=256, L=2.  Encoder runs X[0..255]; decoder runs [X[256], Y[0..254]].
// ---------------------------------------------------------------------------
#define TSTEPS 256      // steps per phase
#define BB     256      // batch
#define DD     131      // input / output feature dim
#define HH     256      // hidden
#define DP     160      // D padded to multiple of 32 (K for layer-0 GEMM)
#define OUTP   144      // 131 padded to multiple of 16 (emit N dim)
#define NWG    64       // persistent workgroups (64 x 4 wave32 = 256 waves)
#define NWAVE  4

typedef unsigned short u16;
typedef unsigned int   u32;
typedef __attribute__((ext_vector_type(16))) __bf16 v16bf;
typedef __attribute__((ext_vector_type(8)))  float  v8f;

union BF16Tile { v16bf v; uint4 q[2]; u16 s[16]; };

__device__ __forceinline__ u16 f2bf(float f) {
  union { float f; u32 u; } x; x.f = f;
  u32 u = x.u;
  u += 0x7fffu + ((u >> 16) & 1u);   // round-to-nearest-even
  return (u16)(u >> 16);
}

__device__ __forceinline__ v8f v8f_zero() {
  v8f z;
#pragma unroll
  for (int i = 0; i < 8; ++i) z[i] = 0.0f;
  return z;
}

__device__ __forceinline__ v8f wmma_bf16(v16bf a, v16bf b, v8f c) {
  // D(f32 16x16) = A(bf16 16x32) * B(bf16 32x16) + C
  return __builtin_amdgcn_wmma_f32_16x16x32_bf16(
      /*neg_a=*/false, a, /*neg_b=*/false, b,
      /*c_mod=*/(short)0, c, /*reuse_a=*/false, /*reuse_b=*/false);
}

// Fast transcendentals: gfx1250 has V_TANH_F32; rcp avoids div expansion.
__device__ __forceinline__ float fast_tanh(float x) {
#if __has_builtin(__builtin_amdgcn_tanhf)
  return __builtin_amdgcn_tanhf(x);
#else
  return tanhf(x);
#endif
}
__device__ __forceinline__ float fast_sigmoid(float x) {
  return __builtin_amdgcn_rcpf(1.0f + __expf(-x));
}

// A tile (16x32 bf16, row-major source, stride ld elements).
// ISA 16-bit A layout: lanes0-15 row=lane, K kb..kb+7 (q0) & kb+16..kb+23 (q1)
// with kb = kc*32; lanes16-31 same rows at kb+8.
__device__ __forceinline__ v16bf ldA_bf(const u16* base, int ld, int m, int kb) {
  BF16Tile t;
  const u16* p = base + (size_t)m * ld + kb;
  t.q[0] = *(const uint4*)(p);
  t.q[1] = *(const uint4*)(p + 16);
  return t.v;
}

// B tile (32x16): column n of B == weight row n ([4H][K] row-major), each
// lane holds K k0..k0+7 (q0) & k0+16..k0+23 (q1) with k0=kc*32,
// lanes16-31 at k0+8 (same layout as A).
__device__ __forceinline__ v16bf ldB_bf(const u16* w, int ld, int n, int k0) {
  BF16Tile t;
  const u16* p = w + (size_t)n * ld + k0;
  t.q[0] = *(const uint4*)(p);
  t.q[1] = *(const uint4*)(p + 16);
  return t.v;
}

// Device-scope generation barrier across the persistent grid.
__device__ __forceinline__ void grid_barrier(u32* cnt, u32* gen) {
  __threadfence();
  __syncthreads();
  if (threadIdx.x == 0) {
    u32 g = __hip_atomic_load(gen, __ATOMIC_RELAXED, __HIP_MEMORY_SCOPE_AGENT);
    u32 a = __hip_atomic_fetch_add(cnt, 1u, __ATOMIC_ACQ_REL, __HIP_MEMORY_SCOPE_AGENT);
    if (a == NWG - 1) {
      __hip_atomic_store(cnt, 0u, __ATOMIC_RELAXED, __HIP_MEMORY_SCOPE_AGENT);
      __hip_atomic_fetch_add(gen, 1u, __ATOMIC_ACQ_REL, __HIP_MEMORY_SCOPE_AGENT);
    } else {
      while (__hip_atomic_load(gen, __ATOMIC_ACQUIRE, __HIP_MEMORY_SCOPE_AGENT) == g)
        __builtin_amdgcn_s_sleep(1);
    }
  }
  __syncthreads();
}

// One LSTM cell step for this wave's 16x16 (rowbase,colbase) h-tile.
// x part: NKX K-chunks with leading dim ldx; h part: 8 chunks (K=256).
// All operands are plain b128 loads matching the WMMA VGPR striping.
template <int NKX>
__device__ __forceinline__ void lstm_tile(
    const u16* xb, int ldx,
    const u16* Wx, const u16* Wh,
    float bi, float bf_, float bg, float bo,
    const u16* hprev, u16* hout, v8f& c,
    int rowbase, int colbase, int lane)
{
  const int half = lane >> 4;
  const int nn   = lane & 15;
  const int m    = rowbase + nn;   // A-matrix row (batch index)
  const int r    = colbase + nn;   // weight row within each gate block

  v8f zi = v8f_zero(), zf = v8f_zero(), zg = v8f_zero(), zo = v8f_zero();

#pragma unroll
  for (int kc = 0; kc < NKX; ++kc) {
    v16bf a = ldA_bf(xb, ldx, m, kc * 32 + half * 8);
    const int k0 = kc * 32 + half * 8;
    zi = wmma_bf16(a, ldB_bf(Wx, ldx, 0 * HH + r, k0), zi);
    zf = wmma_bf16(a, ldB_bf(Wx, ldx, 1 * HH + r, k0), zf);
    zg = wmma_bf16(a, ldB_bf(Wx, ldx, 2 * HH + r, k0), zg);
    zo = wmma_bf16(a, ldB_bf(Wx, ldx, 3 * HH + r, k0), zo);
  }
#pragma unroll
  for (int kc = 0; kc < HH / 32; ++kc) {
    v16bf a = ldA_bf(hprev, HH, m, kc * 32 + half * 8);
    const int k0 = kc * 32 + half * 8;
    zi = wmma_bf16(a, ldB_bf(Wh, HH, 0 * HH + r, k0), zi);
    zf = wmma_bf16(a, ldB_bf(Wh, HH, 1 * HH + r, k0), zf);
    zg = wmma_bf16(a, ldB_bf(Wh, HH, 2 * HH + r, k0), zg);
    zo = wmma_bf16(a, ldB_bf(Wh, HH, 3 * HH + r, k0), zo);
  }
  // C/D layout: vgpr e -> M = e + 8*half, N = lane&15.
#pragma unroll
  for (int e = 0; e < 8; ++e) {
    float I = fast_sigmoid(zi[e] + bi);
    float F = fast_sigmoid(zf[e] + bf_);
    float G = fast_tanh(zg[e] + bg);
    float O = fast_sigmoid(zo[e] + bo);
    float cn = F * c[e] + I * G;
    c[e] = cn;
    float hn = O * fast_tanh(cn);
    int mm = rowbase + e + half * 8;
    hout[(size_t)mm * HH + colbase + nn] = f2bf(hn);
  }
}

struct Params {
  const u16* Xbf;  // [T_IN=257][BB][DP] bf16, K-padded
  const u16* Ybf;  // [TSTEPS][BB][DP]  bf16, K-padded (row 255 unused)
  const u16 *eWx0, *eWh0, *eWx1, *eWh1;
  const u16 *dWx0, *dWh0, *dWx1, *dWh1;
  const u16 *linW;
  const float *ebih0, *ebhh0, *ebih1, *ebhh1;
  const float *dbih0, *dbhh0, *dbih1, *dbhh1;
  const float *linb;
  u16* h0;        // [2][BB*HH] parity double buffer, layer 0
  u16* h1;        // [2][BB*HH] layer 1
  float* logits;  // [BB][OUTP]
  float* out;     // [TSTEPS][BB][DD]
  u32* bcnt; u32* bgen;
};

__global__ __launch_bounds__(NWAVE * 32)
void seq2seq_persistent(Params P)
{
  const int lane = threadIdx.x & 31;
  const int w    = blockIdx.x * NWAVE + (threadIdx.x >> 5);  // 0..255
  const int rowbase = (w >> 4) * 16;   // batch-row tile
  const int colbase = (w & 15) * 16;   // hidden-col tile
  const int cg = colbase + (lane & 15);

  // Hoist combined biases (per-lane scalar: gate column fixed per lane).
  const float eb0i = P.ebih0[0*HH+cg] + P.ebhh0[0*HH+cg];
  const float eb0f = P.ebih0[1*HH+cg] + P.ebhh0[1*HH+cg];
  const float eb0g = P.ebih0[2*HH+cg] + P.ebhh0[2*HH+cg];
  const float eb0o = P.ebih0[3*HH+cg] + P.ebhh0[3*HH+cg];
  const float eb1i = P.ebih1[0*HH+cg] + P.ebhh1[0*HH+cg];
  const float eb1f = P.ebih1[1*HH+cg] + P.ebhh1[1*HH+cg];
  const float eb1g = P.ebih1[2*HH+cg] + P.ebhh1[2*HH+cg];
  const float eb1o = P.ebih1[3*HH+cg] + P.ebhh1[3*HH+cg];
  const float db0i = P.dbih0[0*HH+cg] + P.dbhh0[0*HH+cg];
  const float db0f = P.dbih0[1*HH+cg] + P.dbhh0[1*HH+cg];
  const float db0g = P.dbih0[2*HH+cg] + P.dbhh0[2*HH+cg];
  const float db0o = P.dbih0[3*HH+cg] + P.dbhh0[3*HH+cg];
  const float db1i = P.dbih1[0*HH+cg] + P.dbhh1[0*HH+cg];
  const float db1f = P.dbih1[1*HH+cg] + P.dbhh1[1*HH+cg];
  const float db1g = P.dbih1[2*HH+cg] + P.dbhh1[2*HH+cg];
  const float db1o = P.dbih1[3*HH+cg] + P.dbhh1[3*HH+cg];

  v8f c0 = v8f_zero(), c1 = v8f_zero();  // cell state slices live in VGPRs

  // ---------------- Encoder: 256 steps over X[0..255] ----------------
  for (int t = 0; t < TSTEPS; ++t) {
    const int p = t & 1;
    const u16* xin = P.Xbf + (size_t)t * BB * DP;
    // Prefetch next step's input rows for this wave's row-tile.
    if (lane < 16)
      __builtin_prefetch(P.Xbf + (size_t)(t + 1) * BB * DP +
                         (size_t)(rowbase + lane) * DP, 0, 1);
    lstm_tile<DP / 32>(xin, DP,
        P.eWx0, P.eWh0, eb0i, eb0f, eb0g, eb0o,
        P.h0 + (size_t)(1 - p) * BB * HH, P.h0 + (size_t)p * BB * HH,
        c0, rowbase, colbase, lane);
    grid_barrier(P.bcnt, P.bgen);
    lstm_tile<HH / 32>(P.h0 + (size_t)p * BB * HH, HH,
        P.eWx1, P.eWh1, eb1i, eb1f, eb1g, eb1o,
        P.h1 + (size_t)(1 - p) * BB * HH, P.h1 + (size_t)p * BB * HH,
        c1, rowbase, colbase, lane);
    grid_barrier(P.bcnt, P.bgen);
  }

  // ---------------- Decoder: 256 steps, teacher forcing + emit -------
  for (int td = 0; td < TSTEPS; ++td) {
    const int t = TSTEPS + td;
    const int p = t & 1;
    const u16* xin = (td == 0) ? (P.Xbf + (size_t)TSTEPS * BB * DP)
                               : (P.Ybf + (size_t)(td - 1) * BB * DP);
    if (lane < 16 && td + 1 < TSTEPS)
      __builtin_prefetch(P.Ybf + (size_t)td * BB * DP +
                         (size_t)(rowbase + lane) * DP, 0, 1);
    lstm_tile<DP / 32>(xin, DP,
        P.dWx0, P.dWh0, db0i, db0f, db0g, db0o,
        P.h0 + (size_t)(1 - p) * BB * HH, P.h0 + (size_t)p * BB * HH,
        c0, rowbase, colbase, lane);
    grid_barrier(P.bcnt, P.bgen);
    lstm_tile<HH / 32>(P.h0 + (size_t)p * BB * HH, HH,
        P.dWx1, P.dWh1, db1i, db1f, db1g, db1o,
        P.h1 + (size_t)(1 - p) * BB * HH, P.h1 + (size_t)p * BB * HH,
        c1, rowbase, colbase, lane);
    grid_barrier(P.bcnt, P.bgen);

    // Emit logits = h1 @ linW^T + b : 16 row-tiles x 9 col-tiles = 144 tiles.
    if (w < 144) {
      const int lrb = (w / 9) * 16;
      const int lcb = (w % 9) * 16;
      const int half = lane >> 4;
      const int nn = lane & 15;
      const int m = lrb + nn;
      const u16* hsrc = P.h1 + (size_t)p * BB * HH;
      v8f acc = v8f_zero();
#pragma unroll
      for (int kc = 0; kc < HH / 32; ++kc) {
        v16bf a = ldA_bf(hsrc, HH, m, kc * 32 + half * 8);
        acc = wmma_bf16(a, ldB_bf(P.linW, HH, lcb + nn, kc * 32 + half * 8), acc);
      }
      const int col = lcb + nn;
      const float bb = (col < DD) ? P.linb[col] : 0.0f;
#pragma unroll
      for (int e = 0; e < 8; ++e) {
        int mm = lrb + e + half * 8;
        P.logits[(size_t)mm * OUTP + col] = acc[e] + bb;
      }
    }
    grid_barrier(P.bcnt, P.bgen);

    // Row-wise log-softmax: wave w owns batch row w (32-lane reduction).
    {
      const float* lrow = P.logits + (size_t)w * OUTP;
      float mx = -3.0e38f;
      for (int cidx = lane; cidx < DD; cidx += 32) mx = fmaxf(mx, lrow[cidx]);
#pragma unroll
      for (int o = 16; o; o >>= 1) mx = fmaxf(mx, __shfl_xor(mx, o, 32));
      float s = 0.0f;
      for (int cidx = lane; cidx < DD; cidx += 32) s += __expf(lrow[cidx] - mx);
#pragma unroll
      for (int o = 16; o; o >>= 1) s += __shfl_xor(s, o, 32);
      const float lse = mx + __logf(s);
      float* orow = P.out + ((size_t)td * BB + w) * DD;
      for (int cidx = lane; cidx < DD; cidx += 32) orow[cidx] = lrow[cidx] - lse;
    }
    // No extra barrier needed: the next writes to shared buffers occur only
    // after two more barriers, which imply every wave finished softmax.
  }
}

// ------------------------- preamble kernels --------------------------------
__global__ void zero_u32_kernel(u32* p, int n) {
  int i = blockIdx.x * blockDim.x + threadIdx.x;
  if (i < n) p[i] = 0u;
}

// Row-major f32 [R][C] -> bf16 [Rp][Cp], zero padded.
__global__ void cvt_bf16_pad(const float* __restrict__ src, u16* __restrict__ dst,
                             int R, int C, int Rp, int Cp) {
  int i = blockIdx.x * blockDim.x + threadIdx.x;
  if (i >= Rp * Cp) return;
  int r = i / Cp, cidx = i % Cp;
  float v = (r < R && cidx < C) ? src[(size_t)r * C + cidx] : 0.0f;
  dst[i] = f2bf(v);
}

// ---------------------------------------------------------------------------
extern "C" void kernel_launch(void* const* d_in, const int* in_sizes, int n_in,
                              void* d_out, int out_size, void* d_ws, size_t ws_size,
                              hipStream_t stream) {
  (void)in_sizes; (void)n_in; (void)out_size; (void)ws_size;
  const float* X     = (const float*)d_in[0];
  const float* Y     = (const float*)d_in[1];
  const float* eWih0 = (const float*)d_in[2];
  const float* eWhh0 = (const float*)d_in[3];
  const float* ebih0 = (const float*)d_in[4];
  const float* ebhh0 = (const float*)d_in[5];
  const float* eWih1 = (const float*)d_in[6];
  const float* eWhh1 = (const float*)d_in[7];
  const float* ebih1 = (const float*)d_in[8];
  const float* ebhh1 = (const float*)d_in[9];
  const float* dWih0 = (const float*)d_in[10];
  const float* dWhh0 = (const float*)d_in[11];
  const float* dbih0 = (const float*)d_in[12];
  const float* dbhh0 = (const float*)d_in[13];
  const float* dWih1 = (const float*)d_in[14];
  const float* dWhh1 = (const float*)d_in[15];
  const float* dbih1 = (const float*)d_in[16];
  const float* dbhh1 = (const float*)d_in[17];
  const float* linW  = (const float*)d_in[18];
  const float* linb  = (const float*)d_in[19];

  char* ws = (char*)d_ws;
  size_t o = 0;
  u32* bcnt = (u32*)(ws);            // 2 x u32 barrier state
  u32* bgen = bcnt + 1;
  o += 256;
  u16* h0 = (u16*)(ws + o);  o += (size_t)2 * BB * HH * sizeof(u16);
  u16* h1 = (u16*)(ws + o);  o += (size_t)2 * BB * HH * sizeof(u16);
  const size_t zero_bytes = o;       // barrier + both h double-buffers
  float* logits = (float*)(ws + o);  o += (size_t)BB * OUTP * sizeof(float);
  u16* eWx0b = (u16*)(ws + o); o += (size_t)4 * HH * DP * sizeof(u16);
  u16* eWh0b = (u16*)(ws + o); o += (size_t)4 * HH * HH * sizeof(u16);
  u16* eWx1b = (u16*)(ws + o); o += (size_t)4 * HH * HH * sizeof(u16);
  u16* eWh1b = (u16*)(ws + o); o += (size_t)4 * HH * HH * sizeof(u16);
  u16* dWx0b = (u16*)(ws + o); o += (size_t)4 * HH * DP * sizeof(u16);
  u16* dWh0b = (u16*)(ws + o); o += (size_t)4 * HH * HH * sizeof(u16);
  u16* dWx1b = (u16*)(ws + o); o += (size_t)4 * HH * HH * sizeof(u16);
  u16* dWh1b = (u16*)(ws + o); o += (size_t)4 * HH * HH * sizeof(u16);
  u16* linWb = (u16*)(ws + o); o += (size_t)OUTP * HH * sizeof(u16);
  u16* Xbf   = (u16*)(ws + o); o += (size_t)(TSTEPS + 1) * BB * DP * sizeof(u16);
  u16* Ybf   = (u16*)(ws + o); o += (size_t)TSTEPS * BB * DP * sizeof(u16);

  // Zero barrier state + h buffers (initial hidden state = 0).
  {
    int n = (int)(zero_bytes / 4);
    zero_u32_kernel<<<(n + 255) / 256, 256, 0, stream>>>((u32*)ws, n);
  }
  // Convert weights + activations to bf16 (K-padded / N-padded), once.
  {
    int n;
    n = 4 * HH * DP;
    cvt_bf16_pad<<<(n + 255) / 256, 256, 0, stream>>>(eWih0, eWx0b, 4 * HH, DD, 4 * HH, DP);
    cvt_bf16_pad<<<(n + 255) / 256, 256, 0, stream>>>(dWih0, dWx0b, 4 * HH, DD, 4 * HH, DP);
    n = 4 * HH * HH;
    cvt_bf16_pad<<<(n + 255) / 256, 256, 0, stream>>>(eWhh0, eWh0b, 4 * HH, HH, 4 * HH, HH);
    cvt_bf16_pad<<<(n + 255) / 256, 256, 0, stream>>>(eWih1, eWx1b, 4 * HH, HH, 4 * HH, HH);
    cvt_bf16_pad<<<(n + 255) / 256, 256, 0, stream>>>(eWhh1, eWh1b, 4 * HH, HH, 4 * HH, HH);
    cvt_bf16_pad<<<(n + 255) / 256, 256, 0, stream>>>(dWhh0, dWh0b, 4 * HH, HH, 4 * HH, HH);
    cvt_bf16_pad<<<(n + 255) / 256, 256, 0, stream>>>(dWih1, dWx1b, 4 * HH, HH, 4 * HH, HH);
    cvt_bf16_pad<<<(n + 255) / 256, 256, 0, stream>>>(dWhh1, dWh1b, 4 * HH, HH, 4 * HH, HH);
    n = OUTP * HH;
    cvt_bf16_pad<<<(n + 255) / 256, 256, 0, stream>>>(linW, linWb, DD, HH, OUTP, HH);
    // X: (257*256) rows of 131 -> padded 160.  Y: (256*256) rows.
    n = (TSTEPS + 1) * BB * DP;
    cvt_bf16_pad<<<(n + 255) / 256, 256, 0, stream>>>(X, Xbf, (TSTEPS + 1) * BB, DD,
                                                      (TSTEPS + 1) * BB, DP);
    n = TSTEPS * BB * DP;
    cvt_bf16_pad<<<(n + 255) / 256, 256, 0, stream>>>(Y, Ybf, TSTEPS * BB, DD,
                                                      TSTEPS * BB, DP);
  }

  Params P;
  P.Xbf = Xbf; P.Ybf = Ybf;
  P.eWx0 = eWx0b; P.eWh0 = eWh0b; P.eWx1 = eWx1b; P.eWh1 = eWh1b;
  P.dWx0 = dWx0b; P.dWh0 = dWh0b; P.dWx1 = dWx1b; P.dWh1 = dWh1b;
  P.linW = linWb;
  P.ebih0 = ebih0; P.ebhh0 = ebhh0; P.ebih1 = ebih1; P.ebhh1 = ebhh1;
  P.dbih0 = dbih0; P.dbhh0 = dbhh0; P.dbih1 = dbih1; P.dbhh1 = dbhh1;
  P.linb = linb;
  P.h0 = h0; P.h1 = h1; P.logits = logits;
  P.out = (float*)d_out;
  P.bcnt = bcnt; P.bgen = bgen;

  seq2seq_persistent<<<NWG, NWAVE * 32, 0, stream>>>(P);
}
